// CaduceusSparseMoeBlock_49486613184659
// MI455X (gfx1250) — matrix operand order repaired and checked by reading, not compiled, not run on hardware
//
#include <hip/hip_runtime.h>

// ---------------------------------------------------------------------------
// MoE (8 experts, top-2, SwiGLU) for MI455X / gfx1250.
// Roofline: sparse compute = 0.41 TFLOP (compute-bound vs ~22us of HBM traffic
// at 23.3 TB/s), so we route sparsely and run bf16 WMMA (16x16x32) GEMMs.
// Per-expert bf16 weights (25 MB) fit the 192 MB L2 -> weight reuse across
// M-tiles is served by L2, not HBM. Workspace use: ~286 MB.
// ---------------------------------------------------------------------------

typedef __attribute__((ext_vector_type(16))) __bf16 v16bf;
typedef __attribute__((ext_vector_type(8)))  __bf16 v8bf;
typedef __attribute__((ext_vector_type(4)))  __bf16 v4bf;
typedef __attribute__((ext_vector_type(8)))  float  v8f;

static constexpr int T  = 4 * 2048;   // tokens
static constexpr int Dm = 1024;       // model dim
static constexpr int Fm = 4096;       // ffn dim
static constexpr int NE = 8;          // experts
static constexpr int MT = 32;         // tokens per block (2 WMMA row-blocks)
static constexpr int FT = 128;        // F tile per iteration
static constexpr int XPAD = 8;        // LDS padding (bf16 elems) vs bank conflicts
static constexpr int HPAD = 8;

#define WMMA_BF16(a, b, c) \
  __builtin_amdgcn_wmma_f32_16x16x32_bf16(false, (a), false, (b), (short)0, (c), false, false)

// ---------------------------------------------------------------------------
__global__ void moe_zero_counts_kernel(int* __restrict__ counts) {
  if (threadIdx.x < NE) counts[threadIdx.x] = 0;
}

// Generic fp32 -> bf16 conversion, 4 elements/thread, n multiple of 1024.
__global__ void moe_cvt_f32_bf16_kernel(const float* __restrict__ src,
                                        __bf16* __restrict__ dst) {
  size_t i = ((size_t)blockIdx.x * blockDim.x + threadIdx.x) * 4;
  float4 v = *(const float4*)(src + i);
  v4bf o;
  o[0] = (__bf16)v.x; o[1] = (__bf16)v.y; o[2] = (__bf16)v.z; o[3] = (__bf16)v.w;
  *(v4bf*)(dst + i) = o;
}

// ---------------------------------------------------------------------------
// Router: one wave per token. logits = x[t] @ router_w  (D x 8), fp32 softmax,
// top-2 (first occurrence wins ties, matching jax.lax.top_k), scatter into
// per-expert gather lists with atomic cursors.
__global__ __launch_bounds__(256) void moe_router_kernel(
    const float* __restrict__ x, const float* __restrict__ rw,
    int* __restrict__ counts, int* __restrict__ tok_idx,
    float* __restrict__ tok_w, int* __restrict__ tok_slot) {
  __shared__ float rws[Dm * NE];  // 32 KB
  const int tid = threadIdx.x;
  for (int i = tid; i < Dm * NE; i += 256) rws[i] = rw[i];
  __syncthreads();

  const int wv = tid >> 5, lane = tid & 31;
  const int t = blockIdx.x * 8 + wv;

  float acc[NE];
#pragma unroll
  for (int e = 0; e < NE; ++e) acc[e] = 0.f;
  for (int k = lane; k < Dm; k += 32) {
    float xv = x[(size_t)t * Dm + k];
#pragma unroll
    for (int e = 0; e < NE; ++e) acc[e] += xv * rws[k * NE + e];
  }
#pragma unroll
  for (int off = 16; off > 0; off >>= 1)
#pragma unroll
    for (int e = 0; e < NE; ++e) acc[e] += __shfl_xor(acc[e], off, 32);

  if (lane == 0) {
    float m = acc[0];
#pragma unroll
    for (int e = 1; e < NE; ++e) m = fmaxf(m, acc[e]);
    float p[NE], s = 0.f;
#pragma unroll
    for (int e = 0; e < NE; ++e) { p[e] = __expf(acc[e] - m); s += p[e]; }
    float inv = 1.f / s;
#pragma unroll
    for (int e = 0; e < NE; ++e) p[e] *= inv;

    int e1 = 0;
#pragma unroll
    for (int e = 1; e < NE; ++e) if (p[e] > p[e1]) e1 = e;
    int e2 = (e1 == 0) ? 1 : 0;
#pragma unroll
    for (int e = 0; e < NE; ++e) if (e != e1 && p[e] > p[e2]) e2 = e;

    int pos = atomicAdd(&counts[e1], 1);
    tok_idx[e1 * T + pos] = t; tok_w[e1 * T + pos] = p[e1]; tok_slot[e1 * T + pos] = 0;
    pos = atomicAdd(&counts[e2], 1);
    tok_idx[e2 * T + pos] = t; tok_w[e2 * T + pos] = p[e2]; tok_slot[e2 * T + pos] = 1;
  }
}

// ---------------------------------------------------------------------------
// A-fragment loader (16-bit A 16x32, wave32): lane holds row (lane&15); K is
// two contiguous 8-halfword chunks selected by lane>=16 (ISA 7.12.2).
__device__ __forceinline__ v16bf load_a_frag(const __bf16* rowptr, int k0, int half) {
  v8bf lo = *(const v8bf*)(rowptr + k0 + 8 * half);
  v8bf hi = *(const v8bf*)(rowptr + k0 + 16 + 8 * half);
  v16bf a;
#pragma unroll
  for (int i = 0; i < 8; ++i) { a[i] = lo[i]; a[i + 8] = hi[i]; }
  return a;
}

// Fused expert FFN: per block = (expert, 32-token M-tile).
//   G = Xe @ Wg, U = Xe @ Wu  (F tiled by 128; one 16x16 tile per wave)
//   H = silu(G) * U  -> LDS (bf16)
//   Y += H @ Wd      (each wave owns 128 output columns)
// Epilogue: partial[slot][token] = route_w * Y   (no atomics -> deterministic)
__global__ __launch_bounds__(256) void moe_expert_kernel(
    const __bf16* __restrict__ X16, const __bf16* __restrict__ Wg,
    const __bf16* __restrict__ Wu, const __bf16* __restrict__ Wd,
    const int* __restrict__ counts, const int* __restrict__ tok_idx,
    const float* __restrict__ tok_w, const int* __restrict__ tok_slot,
    float* __restrict__ partial) {
  const int e = blockIdx.y;
  const int nt = counts[e];
  const int base = blockIdx.x * MT;
  if (base >= nt) return;  // whole block exits -> EXEC all-ones at every WMMA

  __shared__ __bf16 Xs[MT][Dm + XPAD];  // 66 KB, row stride 2064 B (16B aligned)
  __shared__ __bf16 Hs[MT][FT + HPAD];  // 8.7 KB, row stride 272 B
  __shared__ int   tks[MT];
  __shared__ float wks[MT];
  __shared__ int   sls[MT];

  const int tid = threadIdx.x;
  if (tid < MT) {
    int r = base + tid;
    if (r < nt) {
      tks[tid] = tok_idx[e * T + r];
      wks[tid] = tok_w[e * T + r];
      sls[tid] = tok_slot[e * T + r];
    } else {
      tks[tid] = -1; wks[tid] = 0.f; sls[tid] = 0;
    }
  }
  __syncthreads();

  // Stage gathered X rows (bf16) into LDS, 16B per op.
  for (int i = tid; i < MT * (Dm / 8); i += 256) {
    int r = i / (Dm / 8), c = (i % (Dm / 8)) * 8;
    v8bf val = {};
    int t = tks[r];
    if (t >= 0) val = *(const v8bf*)(X16 + (size_t)t * Dm + c);
    *(v8bf*)(&Xs[r][c]) = val;
  }
  __syncthreads();

  const int wv = tid >> 5, lane = tid & 31;
  const int half = lane >> 4, l16 = lane & 15;
  const int dbase = wv * 128;  // this wave's output-column slice

  v8f Y[2][8];
#pragma unroll
  for (int rb = 0; rb < 2; ++rb)
#pragma unroll
    for (int n8 = 0; n8 < 8; ++n8) Y[rb][n8] = (v8f)0.f;

  for (int ft = 0; ft < Fm / FT; ++ft) {
    const int fc = ft * FT + wv * 16;  // wave's gate/up column base

    // Hint the next F-tile's weight rows toward L2 (global_prefetch_b8).
    if (ft + 1 < Fm / FT) {
      __builtin_prefetch(Wg + ((size_t)e * Dm + lane) * Fm + fc + FT, 0, 0);
      __builtin_prefetch(Wu + ((size_t)e * Dm + lane) * Fm + fc + FT, 0, 0);
    }

    v8f Ga[2], Ua[2];
#pragma unroll
    for (int rb = 0; rb < 2; ++rb) { Ga[rb] = (v8f)0.f; Ua[rb] = (v8f)0.f; }

    for (int ks = 0; ks < Dm / 32; ++ks) {
      v16bf a0 = load_a_frag(&Xs[l16][0], ks * 32, half);
      v16bf a1 = load_a_frag(&Xs[16 + l16][0], ks * 32, half);
      const int kg = ks * 32 + 8 * half;  // B: lane == N column (l16), K in table order
      v16bf bg, bu;
#pragma unroll
      for (int i = 0; i < 8; ++i) {
        bg[i] = Wg[((size_t)e * Dm + kg + i) * Fm + fc + l16];
        bg[i + 8] = Wg[((size_t)e * Dm + kg + 16 + i) * Fm + fc + l16];
        bu[i] = Wu[((size_t)e * Dm + kg + i) * Fm + fc + l16];
        bu[i + 8] = Wu[((size_t)e * Dm + kg + 16 + i) * Fm + fc + l16];
      }
      Ga[0] = WMMA_BF16(a0, bg, Ga[0]);
      Ga[1] = WMMA_BF16(a1, bg, Ga[1]);
      Ua[0] = WMMA_BF16(a0, bu, Ua[0]);
      Ua[1] = WMMA_BF16(a1, bu, Ua[1]);
    }

    // H = silu(G) * U -> LDS (C/D layout: m = 8*half + vgpr, n = lane&15).
#pragma unroll
    for (int rb = 0; rb < 2; ++rb)
#pragma unroll
      for (int i = 0; i < 8; ++i) {
        float g = Ga[rb][i], u = Ua[rb][i];
        float h = (g / (1.f + __expf(-g))) * u;
        Hs[rb * 16 + 8 * half + i][wv * 16 + l16] = (__bf16)h;
      }
    __syncthreads();

    // Y(32 x 128-per-wave) += H(32 x 128) @ Wd[ft*128 .. , dbase .. dbase+128]
    for (int ks = 0; ks < FT / 32; ++ks) {
      v16bf h0 = load_a_frag(&Hs[l16][0], ks * 32, half);
      v16bf h1 = load_a_frag(&Hs[16 + l16][0], ks * 32, half);
      const int kg = ft * FT + ks * 32 + 8 * half;
#pragma unroll
      for (int n8 = 0; n8 < 8; ++n8) {
        v16bf bd;
#pragma unroll
        for (int i = 0; i < 8; ++i) {
          bd[i] = Wd[((size_t)e * Fm + kg + i) * Dm + dbase + n8 * 16 + l16];
          bd[i + 8] = Wd[((size_t)e * Fm + kg + 16 + i) * Dm + dbase + n8 * 16 + l16];
        }
        Y[0][n8] = WMMA_BF16(h0, bd, Y[0][n8]);
        Y[1][n8] = WMMA_BF16(h1, bd, Y[1][n8]);
      }
    }
    __syncthreads();  // Hs reused next iteration
  }

  // Epilogue: scale by routing weight, scatter rows into partial[slot][token].
#pragma unroll
  for (int rb = 0; rb < 2; ++rb)
#pragma unroll
    for (int i = 0; i < 8; ++i) {
      const int m = rb * 16 + 8 * half + i;
      const int t = tks[m];
      if (t < 0) continue;  // padded row
      const float wt = wks[m];
      const size_t rowoff = ((size_t)sls[m] * T + t) * Dm;
#pragma unroll
      for (int n8 = 0; n8 < 8; ++n8)
        partial[rowoff + dbase + n8 * 16 + l16] = wt * Y[rb][n8][i];
    }
}

// ---------------------------------------------------------------------------
__global__ void moe_combine_kernel(const float* __restrict__ partial,
                                   float* __restrict__ out) {
  size_t i = ((size_t)blockIdx.x * blockDim.x + threadIdx.x) * 4;
  float4 a = *(const float4*)(partial + i);
  float4 b = *(const float4*)(partial + (size_t)T * Dm + i);
  float4 o = make_float4(a.x + b.x, a.y + b.y, a.z + b.z, a.w + b.w);
  *(float4*)(out + i) = o;
}

// ---------------------------------------------------------------------------
extern "C" void kernel_launch(void* const* d_in, const int* in_sizes, int n_in,
                              void* d_out, int out_size, void* d_ws, size_t ws_size,
                              hipStream_t stream) {
  const float* x        = (const float*)d_in[0];  // [T, D]
  const float* router_w = (const float*)d_in[1];  // [D, 8]
  const float* gate_w   = (const float*)d_in[2];  // [8, D, F]
  const float* up_w     = (const float*)d_in[3];  // [8, D, F]
  const float* down_w   = (const float*)d_in[4];  // [8, F, D]
  float* out = (float*)d_out;

  // Workspace carve-up (~286 MB), 256B-aligned sections.
  char* ws = (char*)d_ws;
  size_t off = 0;
  auto carve = [&](size_t bytes) { void* p = ws + off; off += (bytes + 255) & ~size_t(255); return p; };
  __bf16* X16  = (__bf16*)carve((size_t)T * Dm * 2);
  __bf16* Wg16 = (__bf16*)carve((size_t)NE * Dm * Fm * 2);
  __bf16* Wu16 = (__bf16*)carve((size_t)NE * Dm * Fm * 2);
  __bf16* Wd16 = (__bf16*)carve((size_t)NE * Fm * Dm * 2);
  float*  partial = (float*)carve((size_t)2 * T * Dm * 4);
  int*    counts  = (int*)carve(NE * 4);
  int*    tok_idx = (int*)carve((size_t)NE * T * 4);
  float*  tok_w   = (float*)carve((size_t)NE * T * 4);
  int*    tok_slot= (int*)carve((size_t)NE * T * 4);

  moe_zero_counts_kernel<<<1, 32, 0, stream>>>(counts);

  // fp32 -> bf16: activations once, weights once (then served from 192MB L2).
  moe_cvt_f32_bf16_kernel<<<(T * Dm) / 1024, 256, 0, stream>>>(x, X16);
  moe_cvt_f32_bf16_kernel<<<((size_t)NE * Dm * Fm) / 1024, 256, 0, stream>>>(gate_w, Wg16);
  moe_cvt_f32_bf16_kernel<<<((size_t)NE * Dm * Fm) / 1024, 256, 0, stream>>>(up_w, Wu16);
  moe_cvt_f32_bf16_kernel<<<((size_t)NE * Fm * Dm) / 1024, 256, 0, stream>>>(down_w, Wd16);

  moe_router_kernel<<<T / 8, 256, 0, stream>>>(x, router_w, counts, tok_idx, tok_w, tok_slot);

  // expert-major grid: same-expert blocks co-scheduled -> weight reuse in L2.
  moe_expert_kernel<<<dim3(T / MT, NE), 256, 0, stream>>>(
      X16, Wg16, Wu16, Wd16, counts, tok_idx, tok_w, tok_slot, partial);

  moe_combine_kernel<<<(T * Dm) / 1024, 256, 0, stream>>>(partial, out);
}
